// MultiBanditNet_31636729102790
// MI455X (gfx1250) — matrix operand order repaired and checked by reading, not compiled, not run on hardware
//
#include <hip/hip_runtime.h>
#include <cstdint>
#include <cstddef>

// ---------------- problem dims ----------------
#define B_ 1024
#define D_ 512
#define O_ 8
#define A_ 64
#define H_ 1024

typedef __attribute__((ext_vector_type(16))) __bf16 v16bf;
typedef __attribute__((ext_vector_type(8)))  float  v8f;
typedef __attribute__((ext_vector_type(4))) unsigned int v4u;
typedef __attribute__((ext_vector_type(8))) int v8i_;
typedef __attribute__((ext_vector_type(4))) int v4i_;

union Frag { v16bf v; uint4 q[2]; unsigned short u[16]; };
union AccF { v8f   v; float f[8]; };

__device__ __forceinline__ float bf2f(unsigned short u) {
    return __uint_as_float(((unsigned)u) << 16);
}
__device__ __forceinline__ unsigned short f2bf(float f) {
    unsigned u = __float_as_uint(f);
    unsigned r = (u + 0x7FFFu + ((u >> 16) & 1u)) >> 16;   // round-to-nearest-even
    return (unsigned short)r;
}

// ---------------- f32 -> bf16 conversion ----------------
__global__ __launch_bounds__(256)
void cvt_f32_bf16(const float* __restrict__ s, unsigned short* __restrict__ d, int n) {
    int i = blockIdx.x * 256 + threadIdx.x;
    if (i < n) d[i] = f2bf(s[i]);
}

// f32 [Z][K][N] -> bf16 transposed [Z][N][K]  (one-time pass; writes coalesced)
__global__ __launch_bounds__(256)
void cvt_t_f32_bf16(const float* __restrict__ s, unsigned short* __restrict__ d,
                    int K, int N, long long total) {
    long long i = (long long)blockIdx.x * 256 + threadIdx.x;
    if (i >= total) return;
    long long per = (long long)K * N;
    long long z = i / per, r = i - z * per;
    long long n = r / K,  k = r - n * K;
    d[i] = f2bf(s[z * per + k * N + n]);
}

// ---------------- bf16 WMMA GEMM: Y = act(X @ W + b), W pre-transposed ----------------
// A lane: row m=lane%16, two 16B chunks at K = 16*(lane/16) and 16*(lane/16)+8
// B lane: Wt row n=lane%16, one 32B run at K = 16*(lane/16)
// One wave computes a 16x64 strip; all B fragments loaded before the WMMA chain.
__global__ __launch_bounds__(32)
void wmma_gemm_bias_relu(const unsigned short* __restrict__ X,
                         const unsigned short* __restrict__ Wt,
                         const float* __restrict__ bias,
                         unsigned short* __restrict__ Y,
                         int M, int K, int N, int doRelu,
                         long long xz, long long wz, long long bz, long long yz)
{
    const int l   = threadIdx.x;
    const int hf  = l >> 4;
    const int l15 = l & 15;
    const int m0  = blockIdx.x * 16;
    const int n0  = blockIdx.y * 64;
    const int z   = blockIdx.z;

    X  += (long long)z * xz;
    Wt += (long long)z * wz;
    Y  += (long long)z * yz;
    const float* biasz = bias + (long long)z * bz;

    AccF acc[4];
#pragma unroll
    for (int t = 0; t < 4; ++t)
#pragma unroll
        for (int v = 0; v < 8; ++v) acc[t].f[v] = 0.f;

    const unsigned short* xRow = X  + (long long)(m0 + l15) * K;
    const unsigned short* wRow = Wt + (long long)(n0 + l15) * K;

    for (int k0 = 0; k0 < K; k0 += 32) {
        Frag a, b0, b1, b2, b3;
        a.q[0]  = *(const uint4*)(xRow + k0 + 16 * hf);
        a.q[1]  = *(const uint4*)(xRow + k0 + 16 * hf + 8);
        const unsigned short* wp = wRow + k0 + 16 * hf;
        b0.q[0] = *(const uint4*)(wp + 0 * K);
        b0.q[1] = *(const uint4*)(wp + 0 * K + 8);
        b1.q[0] = *(const uint4*)(wp + 16 * K);
        b1.q[1] = *(const uint4*)(wp + 16 * K + 8);
        b2.q[0] = *(const uint4*)(wp + 32 * K);
        b2.q[1] = *(const uint4*)(wp + 32 * K + 8);
        b3.q[0] = *(const uint4*)(wp + 48 * K);
        b3.q[1] = *(const uint4*)(wp + 48 * K + 8);
        acc[0].v = __builtin_amdgcn_wmma_f32_16x16x32_bf16(
            false, a.v, false, b0.v, (short)0, acc[0].v, false, false);
        acc[1].v = __builtin_amdgcn_wmma_f32_16x16x32_bf16(
            false, a.v, false, b1.v, (short)0, acc[1].v, false, false);
        acc[2].v = __builtin_amdgcn_wmma_f32_16x16x32_bf16(
            false, a.v, false, b2.v, (short)0, acc[2].v, false, false);
        acc[3].v = __builtin_amdgcn_wmma_f32_16x16x32_bf16(
            false, a.v, false, b3.v, (short)0, acc[3].v, false, false);
        if (k0 + 32 < K) {   // -> global_prefetch_b8
            __builtin_prefetch(xRow + k0 + 32, 0, 0);
            __builtin_prefetch(wRow + k0 + 32, 0, 0);
        }
    }

#pragma unroll
    for (int t = 0; t < 4; ++t) {
        const int n = n0 + 16 * t + l15;
        const float bv = biasz[n];
#pragma unroll
        for (int v = 0; v < 8; ++v) {
            int m = m0 + v + 8 * hf;            // C/D: M = v + 8*(lane/16)
            float val = acc[t].f[v] + bv;
            if (doRelu) val = fmaxf(val, 0.f);
            Y[(long long)m * N + n] = f2bf(val);
        }
    }
}

// ---------------- q GEMM (N=64) with per-row expert selection scatter ----------------
__global__ __launch_bounds__(32)
void wmma_q_select(const unsigned short* __restrict__ H2,    // [O][B][H] bf16
                   const unsigned short* __restrict__ AWOt,  // [O][A][H] bf16 (transposed)
                   const float* __restrict__ abo,            // [O][A] f32
                   const int* __restrict__ sel,              // [B]
                   float* __restrict__ qsel)                 // [B][A] f32
{
    const int l   = threadIdx.x;
    const int hf  = l >> 4;
    const int l15 = l & 15;
    const int m0  = blockIdx.x * 16;
    const int z   = blockIdx.z;

    const unsigned short* X  = H2   + (long long)z * B_ * H_;
    const unsigned short* Wt = AWOt + (long long)z * A_ * H_;

    AccF acc[4];
#pragma unroll
    for (int t = 0; t < 4; ++t)
#pragma unroll
        for (int v = 0; v < 8; ++v) acc[t].f[v] = 0.f;

    const unsigned short* xRow = X  + (long long)(m0 + l15) * H_;
    const unsigned short* wRow = Wt + (long long)l15 * H_;

    for (int k0 = 0; k0 < H_; k0 += 32) {
        Frag a, b0, b1, b2, b3;
        a.q[0]  = *(const uint4*)(xRow + k0 + 16 * hf);
        a.q[1]  = *(const uint4*)(xRow + k0 + 16 * hf + 8);
        const unsigned short* wp = wRow + k0 + 16 * hf;
        b0.q[0] = *(const uint4*)(wp + 0 * H_);
        b0.q[1] = *(const uint4*)(wp + 0 * H_ + 8);
        b1.q[0] = *(const uint4*)(wp + 16 * H_);
        b1.q[1] = *(const uint4*)(wp + 16 * H_ + 8);
        b2.q[0] = *(const uint4*)(wp + 32 * H_);
        b2.q[1] = *(const uint4*)(wp + 32 * H_ + 8);
        b3.q[0] = *(const uint4*)(wp + 48 * H_);
        b3.q[1] = *(const uint4*)(wp + 48 * H_ + 8);
        acc[0].v = __builtin_amdgcn_wmma_f32_16x16x32_bf16(
            false, a.v, false, b0.v, (short)0, acc[0].v, false, false);
        acc[1].v = __builtin_amdgcn_wmma_f32_16x16x32_bf16(
            false, a.v, false, b1.v, (short)0, acc[1].v, false, false);
        acc[2].v = __builtin_amdgcn_wmma_f32_16x16x32_bf16(
            false, a.v, false, b2.v, (short)0, acc[2].v, false, false);
        acc[3].v = __builtin_amdgcn_wmma_f32_16x16x32_bf16(
            false, a.v, false, b3.v, (short)0, acc[3].v, false, false);
        if (k0 + 32 < H_) {
            __builtin_prefetch(xRow + k0 + 32, 0, 0);
            __builtin_prefetch(wRow + k0 + 32, 0, 0);
        }
    }

#pragma unroll
    for (int t = 0; t < 4; ++t) {
        const int n = 16 * t + l15;
        const float bv = abo[(long long)z * A_ + n];
#pragma unroll
        for (int v = 0; v < 8; ++v) {
            int m = m0 + v + 8 * hf;
            if (sel[m] == z)
                qsel[(long long)m * A_ + n] = acc[t].f[v] + bv;
        }
    }
}

// ---------------- TDM probe: tensor_load_to_lds of a 16x64 bf16 tile ----------------
// D# packed per CDNA5 ISA ch.8 (group0: count/lds_addr/global_addr/type,
// group1: data_size/tensor dims/tile dims/strides). clang-23 6-arg builtin form.
__global__ __launch_bounds__(32)
void tdm_stage_kernel(const unsigned short* __restrict__ src,  // bf16 tensor [1024][512]
                      unsigned short* __restrict__ dst)        // scratch [16*64]
{
    __shared__ unsigned short tile[16 * 64];
    unsigned long long ga = (unsigned long long)(uintptr_t)src;
    unsigned ldsa = (unsigned)(uintptr_t)&tile[0];

    v4u g0;
    g0[0] = 1u;                                        // count=1 (valid user D#)
    g0[1] = ldsa;                                      // lds_addr
    g0[2] = (unsigned)(ga & 0xFFFFFFFFu);              // global_addr[31:0]
    g0[3] = (unsigned)((ga >> 32) & 0x01FFFFFFu)       // global_addr[56:32]
          | 0x80000000u;                               // type=2 ("image") at [127:126]
    v8i_ g1;
    g1[0] = (int)(1u << 16);        // data_size=1 -> 2 bytes/elem
    g1[1] = (int)(512u << 16);      // tensor_dim0 lo16 (=512)
    g1[2] = (int)(1024u << 16);     // tensor_dim0 hi16=0 | tensor_dim1 lo16 (=1024)
    g1[3] = (int)(64u << 16);       // tensor_dim1 hi16=0 | tile_dim0=64
    g1[4] = 16;                     // tile_dim1=16 | tile_dim2=0
    g1[5] = 512;                    // tensor_dim0_stride lo32
    g1[6] = 0;                      // tensor_dim0_stride hi16 | tensor_dim1_stride lo16
    g1[7] = 0;                      // tensor_dim1_stride hi32
    v4i_ zero4 = {0, 0, 0, 0};
    v8i_ zero8 = {0, 0, 0, 0, 0, 0, 0, 0};

    __builtin_amdgcn_tensor_load_to_lds(g0, g1, zero4, zero4, zero8, 0);
    __builtin_amdgcn_s_wait_tensorcnt(0);
    __syncthreads();
    for (int i = threadIdx.x; i < 16 * 64; i += 32) dst[i] = tile[i];
}

// ---------------- threefry2x32 (JAX-compatible) ----------------
__device__ __forceinline__ unsigned rotl32(unsigned x, int r) {
    return (x << r) | (x >> (32 - r));
}
__device__ void threefry2x32(unsigned ks0, unsigned ks1,
                             unsigned x0, unsigned x1,
                             unsigned& o0, unsigned& o1)
{
    const unsigned ks2 = ks0 ^ ks1 ^ 0x1BD11BDAu;
    const int R0[4] = {13, 15, 26, 6};
    const int R1[4] = {17, 29, 16, 24};
    x0 += ks0; x1 += ks1;
#pragma unroll
    for (int i = 0; i < 4; ++i) { x0 += x1; x1 = rotl32(x1, R0[i]); x1 ^= x0; }
    x0 += ks1; x1 += ks2 + 1u;
#pragma unroll
    for (int i = 0; i < 4; ++i) { x0 += x1; x1 = rotl32(x1, R1[i]); x1 ^= x0; }
    x0 += ks2; x1 += ks0 + 2u;
#pragma unroll
    for (int i = 0; i < 4; ++i) { x0 += x1; x1 = rotl32(x1, R0[i]); x1 ^= x0; }
    x0 += ks0; x1 += ks1 + 3u;
#pragma unroll
    for (int i = 0; i < 4; ++i) { x0 += x1; x1 = rotl32(x1, R1[i]); x1 ^= x0; }
    x0 += ks1; x1 += ks2 + 4u;
#pragma unroll
    for (int i = 0; i < 4; ++i) { x0 += x1; x1 = rotl32(x1, R0[i]); x1 ^= x0; }
    x0 += ks2; x1 += ks0 + 5u;
    o0 = x0; o1 = x1;
}

// ---------------- heads: async-to-LDS row staging + logits/softmax/categorical ----------------
__global__ __launch_bounds__(256)
void heads_kernel(const unsigned short* __restrict__ Hopt,   // [B][H] bf16
                  const unsigned short* __restrict__ Hterm,  // [B][H] bf16
                  const float* __restrict__ ow2, const float* __restrict__ ob2,
                  const float* __restrict__ tw2, const float* __restrict__ tb2,
                  float* __restrict__ out_optp,   // [B][O]
                  float* __restrict__ out_term,   // [B]
                  float* __restrict__ out_oarg,   // [B]
                  int* __restrict__ sel)          // [B]
{
    const int b = blockIdx.x, t = threadIdx.x;

    // Stage this row's Hopt (2KB) + Hterm (2KB) into LDS via async DMA:
    // each of 256 lanes issues one 16B global_load_async_to_lds_b128.
    __shared__ unsigned short rowbuf[2 * H_];
    {
        const unsigned short* gsrc = (t < 128)
            ? (Hopt  + (size_t)b * H_ + (size_t)t * 8)
            : (Hterm + (size_t)b * H_ + (size_t)(t - 128) * 8);
        unsigned ldsoff = (unsigned)(uintptr_t)&rowbuf[t * 8];
        asm volatile("global_load_async_to_lds_b128 %0, %1, off"
                     :: "v"(ldsoff), "v"(gsrc) : "memory");
        asm volatile("s_wait_asynccnt 0x0" ::: "memory");
    }
    __syncthreads();

    float acc[9];
#pragma unroll
    for (int i = 0; i < 9; ++i) acc[i] = 0.f;

    for (int k = t; k < H_; k += 256) {
        float xo = bf2f(rowbuf[k]);
#pragma unroll
        for (int o = 0; o < O_; ++o) acc[o] += xo * ow2[k * O_ + o];
        float xt = bf2f(rowbuf[H_ + k]);
        acc[8] += xt * tw2[k];
    }
#pragma unroll
    for (int i = 0; i < 9; ++i)
        for (int off = 16; off; off >>= 1) acc[i] += __shfl_down(acc[i], off);

    __shared__ float red[8][9];
    const int wid = t >> 5, lane = t & 31;
    if (lane == 0) {
#pragma unroll
        for (int i = 0; i < 9; ++i) red[wid][i] = acc[i];
    }
    __syncthreads();

    if (t == 0) {
        float lg[9];
#pragma unroll
        for (int i = 0; i < 9; ++i) {
            float s = 0.f;
            for (int wv = 0; wv < 8; ++wv) s += red[wv][i];
            lg[i] = s;
        }
        for (int o = 0; o < O_; ++o) lg[o] += ob2[o];
        float tl = lg[8] + tb2[0];
        out_term[b] = 1.f / (1.f + expf(-tl));

        float mx = lg[0]; int am = 0;
        for (int o = 1; o < O_; ++o) if (lg[o] > mx) { mx = lg[o]; am = o; }
        float es = 0.f, e[O_];
        for (int o = 0; o < O_; ++o) { e[o] = expf(lg[o] - mx); es += e[o]; }
        for (int o = 0; o < O_; ++o) out_optp[(size_t)b * O_ + o] = e[o] / es;
        out_oarg[b] = (float)am;

        // jax.random.categorical(key(42), logits): argmax(logits + gumbel)
        float best = -3.4e38f; int bi = 0;
        for (int o = 0; o < O_; ++o) {
            unsigned idx = (unsigned)(b * O_ + o), r0, r1, bits;
            if (idx < 4096u) { threefry2x32(0u, 42u, idx, idx + 4096u, r0, r1); bits = r0; }
            else             { threefry2x32(0u, 42u, idx - 4096u, idx, r0, r1); bits = r1; }
            float f = __uint_as_float((bits >> 9) | 0x3f800000u) - 1.0f;
            const float TINY = 1.17549435e-38f;
            float u = f * (1.0f - TINY) + TINY;
            u = fmaxf(u, TINY);
            float g = -logf(-logf(u));
            float vv = lg[o] + g;
            if (vv > best) { best = vv; bi = o; }
        }
        sel[b] = bi;
    }
}

// ---------------- action softmax + argmax ----------------
__global__ __launch_bounds__(64)
void action_kernel(const float* __restrict__ qsel,
                   float* __restrict__ out_ap,   // [B][A]
                   float* __restrict__ out_sa)   // [B]
{
    const int b = blockIdx.x, t = threadIdx.x;
    __shared__ float q[A_];
    __shared__ float mxs, sums;
    __shared__ int ams;
    q[t] = qsel[(size_t)b * A_ + t];
    __syncthreads();
    if (t == 0) {
        float mx = q[0]; int am = 0;
        for (int a = 1; a < A_; ++a) if (q[a] > mx) { mx = q[a]; am = a; }
        float s = 0.f;
        for (int a = 0; a < A_; ++a) s += expf(q[a] - mx);
        mxs = mx; sums = s; ams = am;
    }
    __syncthreads();
    out_ap[(size_t)b * A_ + t] = expf(q[t] - mxs) / sums;
    if (t == 0) out_sa[b] = (float)ams;
}

// ---------------- workspace layout (all weight mats stored transposed) ----------------
constexpr size_t alg(size_t x) { return (x + 255) & ~(size_t)255; }
constexpr size_t OFF_XB    = 0;
constexpr size_t OFF_OW1T  = OFF_XB    + alg((size_t)B_ * D_ * 2);
constexpr size_t OFF_TW1T  = OFF_OW1T  + alg((size_t)D_ * H_ * 2);
constexpr size_t OFF_AW1T  = OFF_TW1T  + alg((size_t)D_ * H_ * 2);
constexpr size_t OFF_AWHT  = OFF_AW1T  + alg((size_t)O_ * D_ * H_ * 2);
constexpr size_t OFF_AWOT  = OFF_AWHT  + alg((size_t)O_ * H_ * H_ * 2);
constexpr size_t OFF_HOPT  = OFF_AWOT  + alg((size_t)O_ * H_ * A_ * 2);
constexpr size_t OFF_HTERM = OFF_HOPT  + alg((size_t)B_ * H_ * 2);
constexpr size_t OFF_H1    = OFF_HTERM + alg((size_t)B_ * H_ * 2);
constexpr size_t OFF_H2    = OFF_H1    + alg((size_t)O_ * B_ * H_ * 2);
constexpr size_t OFF_SEL   = OFF_H2    + alg((size_t)O_ * B_ * H_ * 2);
constexpr size_t OFF_QSEL  = OFF_SEL   + alg((size_t)B_ * 4);
constexpr size_t OFF_TDMS  = OFF_QSEL  + alg((size_t)B_ * A_ * 4);

extern "C" void kernel_launch(void* const* d_in, const int* in_sizes, int n_in,
                              void* d_out, int out_size, void* d_ws, size_t ws_size,
                              hipStream_t stream)
{
    const float* state = (const float*)d_in[0];
    const float* ow1   = (const float*)d_in[1];
    const float* ob1   = (const float*)d_in[2];
    const float* ow2   = (const float*)d_in[3];
    const float* ob2   = (const float*)d_in[4];
    const float* aw1   = (const float*)d_in[5];
    const float* ab1   = (const float*)d_in[6];
    const float* awh   = (const float*)d_in[7];
    const float* abh   = (const float*)d_in[8];
    const float* awo   = (const float*)d_in[9];
    const float* abo   = (const float*)d_in[10];
    const float* tw1   = (const float*)d_in[11];
    const float* tb1   = (const float*)d_in[12];
    const float* tw2   = (const float*)d_in[13];
    const float* tb2   = (const float*)d_in[14];

    char* w = (char*)d_ws;
    unsigned short* Xb    = (unsigned short*)(w + OFF_XB);
    unsigned short* OW1t  = (unsigned short*)(w + OFF_OW1T);   // [H,D]
    unsigned short* TW1t  = (unsigned short*)(w + OFF_TW1T);   // [H,D]
    unsigned short* AW1t  = (unsigned short*)(w + OFF_AW1T);   // [O][H][D]
    unsigned short* AWHt  = (unsigned short*)(w + OFF_AWHT);   // [O][H][H]
    unsigned short* AWOt  = (unsigned short*)(w + OFF_AWOT);   // [O][A][H]
    unsigned short* Hopt  = (unsigned short*)(w + OFF_HOPT);
    unsigned short* Hterm = (unsigned short*)(w + OFF_HTERM);
    unsigned short* H1w   = (unsigned short*)(w + OFF_H1);
    unsigned short* H2w   = (unsigned short*)(w + OFF_H2);
    int*            SEL   = (int*)(w + OFF_SEL);
    float*          QSEL  = (float*)(w + OFF_QSEL);
    unsigned short* TDMS  = (unsigned short*)(w + OFF_TDMS);

    float* out      = (float*)d_out;
    float* out_optp = out;                         // [B,O]
    float* out_ap   = out + (size_t)B_ * O_;       // [B,A]
    float* out_term = out_ap + (size_t)B_ * A_;    // [B,1]
    float* out_oarg = out_term + B_;               // [B]
    float* out_sa   = out_oarg + B_;               // [B]

    // 1) convert: state plain, all GEMM weights transposed to [N,K]
    cvt_f32_bf16<<<(B_ * D_ + 255) / 256, 256, 0, stream>>>(state, Xb, B_ * D_);
    auto cvtT = [&](const float* s, unsigned short* d, int K, int N, long long z) {
        long long tot = z * (long long)K * N;
        cvt_t_f32_bf16<<<(unsigned)((tot + 255) / 256), 256, 0, stream>>>(s, d, K, N, tot);
    };
    cvtT(ow1, OW1t, D_, H_, 1);
    cvtT(tw1, TW1t, D_, H_, 1);
    cvtT(aw1, AW1t, D_, H_, O_);
    cvtT(awh, AWHt, H_, H_, O_);
    cvtT(awo, AWOt, H_, A_, O_);

    // TDM probe (deterministic; writes only to private scratch slot)
    tdm_stage_kernel<<<1, 32, 0, stream>>>(Xb, TDMS);

    const dim3 wv(32);
    // 2) option hidden: relu(state @ ow1 + ob1)
    wmma_gemm_bias_relu<<<dim3(B_ / 16, H_ / 64, 1), wv, 0, stream>>>(
        Xb, OW1t, ob1, Hopt, B_, D_, H_, 1, 0, 0, 0, 0);
    //    termination hidden
    wmma_gemm_bias_relu<<<dim3(B_ / 16, H_ / 64, 1), wv, 0, stream>>>(
        Xb, TW1t, tb1, Hterm, B_, D_, H_, 1, 0, 0, 0, 0);
    // 3) expert layer 1 (all experts)
    wmma_gemm_bias_relu<<<dim3(B_ / 16, H_ / 64, O_), wv, 0, stream>>>(
        Xb, AW1t, ab1, H1w, B_, D_, H_, 1,
        0, (long long)D_ * H_, H_, (long long)B_ * H_);
    // 4) expert hidden
    wmma_gemm_bias_relu<<<dim3(B_ / 16, H_ / 64, O_), wv, 0, stream>>>(
        H1w, AWHt, abh, H2w, B_, H_, H_, 1,
        (long long)B_ * H_, (long long)H_ * H_, H_, (long long)B_ * H_);
    // 5) heads + threefry-gumbel option selection (async-to-LDS staging)
    heads_kernel<<<B_, 256, 0, stream>>>(
        Hopt, Hterm, ow2, ob2, tw2, tb2, out_optp, out_term, out_oarg, SEL);
    // 6) q GEMM + selection scatter
    wmma_q_select<<<dim3(B_ / 16, 1, O_), wv, 0, stream>>>(
        H2w, AWOt, abo, SEL, QSEL);
    // 7) action softmax + argmax
    action_kernel<<<B_, 64, 0, stream>>>(QSEL, out_ap, out_sa);
}
